// DiscreteVAE_24876450578580
// MI455X (gfx1250) — matrix operand order repaired and checked by reading, not compiled, not run on hardware
//
#include <hip/hip_runtime.h>
#include <hip/hip_bf16.h>
#include <math.h>

typedef __attribute__((ext_vector_type(16))) __bf16 v16bf;
typedef __attribute__((ext_vector_type(8)))  float  v8f;

union Frag {
  uint4 q[2];
  v16bf v;
};

__device__ inline unsigned short f2bf(float f) {
  union { float f; unsigned u; } x; x.f = f;
  unsigned u = x.u;
  u += 0x7FFFu + ((u >> 16) & 1u);   // round-to-nearest-even
  return (unsigned short)(u >> 16);
}

// ---------------------------------------------------------------------------
// Implicit-GEMM conv (NCHW, OIHW weights) via v_wmma_f32_16x16x32_bf16.
// 64(cout) x 64(pix) tile per block, 8 waves, each wave: 16x32 subtile.
// K staged 64-deep per LDS round -> 4 WMMAs per barrier pair.
// Templated on kernel size / fused nearest-x2 upsample so the per-element
// im2col index math uses constant divisions. All pixel indices are 32-bit.
// ---------------------------------------------------------------------------
template <int KS, int UPS>
__global__ __launch_bounds__(256) void conv_wmma_kernel(
    const float* __restrict__ in, const float* __restrict__ wgt,
    const float* __restrict__ bias, const float* __restrict__ residual,
    float* __restrict__ out,
    int B, int Cin, int Hin, int Win, int Cout, int Hout, int Wout,
    int pad, int stride)
{
  constexpr int KSQ = KS * KS;
  __shared__ alignas(16) unsigned short Alds[64 * 72];
  __shared__ alignas(16) unsigned short Blds[64 * 72];
  const int tid  = threadIdx.x;
  const int wave = tid >> 5;
  const int lane = tid & 31;
  const int wm = wave & 3;         // m subtile 0..3
  const int wn = wave >> 2;        // n half 0..1
  const int mblk = blockIdx.y * 64;
  const unsigned HWout = (unsigned)(Hout * Wout);
  const unsigned npix = (unsigned)B * HWout;
  const unsigned nblk = blockIdx.x * 64u;
  const int Ktot = Cin * KSQ;
  const int Hv = UPS ? (Hin << 1) : Hin;   // virtual (upsampled) input dims
  const int Wv = UPS ? (Win << 1) : Win;
  const int HinWin = Hin * Win;

  v8f zero = {0.f,0.f,0.f,0.f,0.f,0.f,0.f,0.f};
  v8f acc[2]; acc[0] = zero; acc[1] = zero;

  // staging coordinates: 4 threads per row, 16 consecutive k each
  const int srow = tid >> 2;
  const int skb  = (tid & 3) * 16;
  const unsigned n_g = nblk + (unsigned)srow;
  int pb = 0, poy = 0, pox = 0;
  const bool nvalid = (n_g < npix);
  if (nvalid) {
    pb = (int)(n_g / HWout);
    unsigned rem = n_g - (unsigned)pb * HWout;
    poy = (int)(rem / (unsigned)Wout);
    pox = (int)(rem - (unsigned)poy * (unsigned)Wout);
  }
  const bool mvalid = (mblk + srow) < Cout;
  const int mg = mblk + srow;

  for (int k0 = 0; k0 < Ktot; k0 += 64) {
    // stage A: weights -> Alds[m][k] (bf16)
    {
      unsigned short* dst = &Alds[srow * 72 + skb];
      #pragma unroll
      for (int j = 0; j < 16; ++j) {
        int k = k0 + skb + j;
        float v = 0.f;
        if (mvalid && k < Ktot) {
          int ci = k / KSQ;                 // constant divisor
          int r  = k - ci * KSQ;
          v = wgt[(mg * Cin + ci) * KSQ + r];
        }
        dst[j] = f2bf(v);
      }
    }
    // stage B: im2col -> Blds[n][k] (bf16), fused upsample mapping
    {
      unsigned short* dst = &Blds[srow * 72 + skb];
      #pragma unroll
      for (int j = 0; j < 16; ++j) {
        int k = k0 + skb + j;
        float v = 0.f;
        if (nvalid && k < Ktot) {
          int ci = k / KSQ;
          int r  = k - ci * KSQ;
          int kh = r / KS, kw = r - kh * KS;
          int iy = poy * stride - pad + kh;
          int ix = pox * stride - pad + kw;
          if ((unsigned)iy < (unsigned)Hv && (unsigned)ix < (unsigned)Wv) {
            int sy = UPS ? (iy >> 1) : iy;
            int sx = UPS ? (ix >> 1) : ix;
            v = in[(pb * Cin + ci) * HinWin + sy * Win + sx];
          }
        }
        dst[j] = f2bf(v);
      }
    }
    __syncthreads();

    #pragma unroll
    for (int c = 0; c < 2; ++c) {
      Frag a, b0, b1;
      const unsigned short* ar = &Alds[(wm * 16 + (lane & 15)) * 72 + c * 32];
      const int kb = (lane < 16) ? 0 : 16;
      a.q[0] = *(const uint4*)(ar + kb);
      a.q[1] = *(const uint4*)(ar + kb + 8);
      const int kc = (lane < 16) ? 0 : 16;
      const unsigned short* br0 = &Blds[(wn * 32 + (lane & 15)) * 72 + c * 32 + kc];
      b0.q[0] = ((const uint4*)br0)[0];
      b0.q[1] = ((const uint4*)br0)[1];
      const unsigned short* br1 = &Blds[(wn * 32 + 16 + (lane & 15)) * 72 + c * 32 + kc];
      b1.q[0] = ((const uint4*)br1)[0];
      b1.q[1] = ((const uint4*)br1)[1];
      acc[0] = __builtin_amdgcn_wmma_f32_16x16x32_bf16(false, a.v, false, b0.v,
                                                       (short)0, acc[0], false, false);
      acc[1] = __builtin_amdgcn_wmma_f32_16x16x32_bf16(false, a.v, false, b1.v,
                                                       (short)0, acc[1], false, false);
    }
    __syncthreads();
  }

  // epilogue: C/D layout -> NCHW store with bias (+residual)
  const int m0 = mblk + wm * 16 + ((lane >> 4) << 3);
  const int ncol = lane & 15;
  #pragma unroll
  for (int s = 0; s < 2; ++s) {
    unsigned n = nblk + (unsigned)(wn * 32 + s * 16 + ncol);
    if (n >= npix) continue;
    unsigned b = n / HWout;
    unsigned rem = n - b * HWout;
    #pragma unroll
    for (int r = 0; r < 8; ++r) {
      int m = m0 + r;
      if (m >= Cout) continue;
      unsigned oi = (b * (unsigned)Cout + (unsigned)m) * HWout + rem;
      float v = acc[s][r] + bias[m];
      if (residual) v += residual[oi];
      out[oi] = v;
    }
  }
}

// ---------------------------------------------------------------------------
// GroupNorm (32 groups, biased variance) with optional SiLU. One block per
// (batch, group). Channel-major write loop avoids per-element division.
// ---------------------------------------------------------------------------
__global__ __launch_bounds__(256) void gn_kernel(
    const float* __restrict__ in, const float* __restrict__ gamma,
    const float* __restrict__ beta, float* __restrict__ out,
    int C, int HW, int silu)
{
  const int grp = blockIdx.x & 31;
  const int b   = blockIdx.x >> 5;
  const int cpg = C / 32;
  const long base = ((long)b * C + (long)grp * cpg) * HW;
  const int n = cpg * HW;
  float s = 0.f, s2 = 0.f;
  for (int i = threadIdx.x; i < n; i += 256) {
    float v = in[base + i];
    s += v; s2 += v * v;
  }
  __shared__ float rs[256], rq[256];
  rs[threadIdx.x] = s; rq[threadIdx.x] = s2;
  __syncthreads();
  for (int o = 128; o > 0; o >>= 1) {
    if (threadIdx.x < o) {
      rs[threadIdx.x] += rs[threadIdx.x + o];
      rq[threadIdx.x] += rq[threadIdx.x + o];
    }
    __syncthreads();
  }
  const float mean = rs[0] / (float)n;
  const float var  = rq[0] / (float)n - mean * mean;
  const float inv  = rsqrtf(var + 1e-5f);
  for (int c = 0; c < cpg; ++c) {
    const float ga = gamma[grp * cpg + c] * inv;
    const float be = beta[grp * cpg + c];
    const long cb = base + (long)c * HW;
    for (int i = threadIdx.x; i < HW; i += 256) {
      float v = (in[cb + i] - mean) * ga + be;
      if (silu) v = v / (1.f + __expf(-v));
      out[cb + i] = v;
    }
  }
}

// channel concat in NCHW (32-bit index math)
__global__ void concat_kernel(const float* __restrict__ a, const float* __restrict__ bb,
                              float* __restrict__ out, int B, int C1, int C2, int HW)
{
  const unsigned Ct = (unsigned)(C1 + C2);
  const unsigned total = (unsigned)B * Ct * (unsigned)HW;
  for (unsigned i = blockIdx.x * blockDim.x + threadIdx.x; i < total;
       i += gridDim.x * blockDim.x) {
    unsigned hw = i % (unsigned)HW;
    unsigned t  = i / (unsigned)HW;
    unsigned c  = t % Ct;
    unsigned b  = t / Ct;
    out[i] = (c < (unsigned)C1)
                 ? a[(b * C1 + c) * HW + hw]
                 : bb[(b * C2 + (c - C1)) * HW + hw];
  }
}

// ---------------------------------------------------------------------------
// QKVAttentionLegacy core: qkv is (B, H*3*ch, T) with per-head [q|k|v]
// interleave. One block per (b, head); LDS softmax row per query t.
// ---------------------------------------------------------------------------
__global__ __launch_bounds__(256) void attn_core_kernel(
    const float* __restrict__ qkv, float* __restrict__ outp,
    int C, int T, int heads)
{
  const int bh = blockIdx.x;
  const int b = bh / heads, h = bh % heads;
  const int ch = C / heads;                       // 16
  const float scale = rsqrtf(sqrtf((float)ch));   // 1/sqrt(sqrt(ch))
  const float* qp = qkv + ((long)b * 3 * C + (long)h * 3 * ch) * T;
  const float* kp = qp + (long)ch * T;
  const float* vp = qp + 2L * ch * T;
  float* op = outp + ((long)b * C + (long)h * ch) * T;
  __shared__ float sc[1024];
  __shared__ float red[256];
  __shared__ float qsh[64];
  const int tid = threadIdx.x;
  for (int t = 0; t < T; ++t) {
    if (tid < ch) qsh[tid] = qp[tid * T + t] * scale;
    __syncthreads();
    for (int s = tid; s < T; s += 256) {
      float d = 0.f;
      for (int c = 0; c < ch; ++c) d += qsh[c] * kp[c * T + s];
      sc[s] = d * scale;
    }
    __syncthreads();
    float m = -3.0e38f;
    for (int s = tid; s < T; s += 256) m = fmaxf(m, sc[s]);
    red[tid] = m; __syncthreads();
    for (int o = 128; o > 0; o >>= 1) {
      if (tid < o) red[tid] = fmaxf(red[tid], red[tid + o]);
      __syncthreads();
    }
    const float mx = red[0];
    __syncthreads();
    float ssum = 0.f;
    for (int s = tid; s < T; s += 256) {
      float e = __expf(sc[s] - mx);
      sc[s] = e; ssum += e;
    }
    red[tid] = ssum; __syncthreads();
    for (int o = 128; o > 0; o >>= 1) {
      if (tid < o) red[tid] += red[tid + o];
      __syncthreads();
    }
    const float inv = 1.f / red[0];
    __syncthreads();
    const int c = tid >> 4, sg = tid & 15;
    float p = 0.f;
    if (c < ch) for (int s = sg; s < T; s += 16) p += sc[s] * vp[c * T + s];
    red[tid] = p; __syncthreads();
    if (tid < ch) {
      float acc = 0.f;
      for (int j = 0; j < 16; ++j) acc += red[tid * 16 + j];
      op[tid * T + t] = acc * inv;
    }
    __syncthreads();
  }
}

// x2[n] = sum_d logits[n,d]^2, logits gathered from encoder output NCHW
__global__ void rowsq_kernel(const float* __restrict__ enc, float* __restrict__ x2,
                             int B, int D, int HW)
{
  unsigned n = blockIdx.x * blockDim.x + threadIdx.x;
  unsigned N = (unsigned)B * (unsigned)HW;
  if (n >= N) return;
  unsigned b = n / (unsigned)HW, rem = n - b * (unsigned)HW;
  const float* p = enc + (size_t)b * D * HW + rem;
  float s = 0.f;
  for (int d = 0; d < D; ++d) { float v = p[(size_t)d * HW]; s += v * v; }
  x2[n] = s;
}

// e2[k] = sum_d embed[d,k]^2
__global__ void colsq_kernel(const float* __restrict__ embed, float* __restrict__ e2,
                             int D, int K)
{
  int k = blockIdx.x * blockDim.x + threadIdx.x;
  if (k >= K) return;
  float s = 0.f;
  for (int d = 0; d < D; ++d) { float v = embed[(size_t)d * K + k]; s += v * v; }
  e2[k] = s;
}

// ---------------------------------------------------------------------------
// VQ: dist = |x|^2 - 2 x.e + |e|^2, argmin over 8192 codes, WMMA for x.e.
// 64 positions per block; D staged 64-deep -> 4 WMMAs per barrier pair.
// Running argmin via packed u64 LDS atomicMin. Writes quantized NCHW tensor
// and accumulates commitment-loss sum.
// ---------------------------------------------------------------------------
__global__ __launch_bounds__(256) void vq_kernel(
    const float* __restrict__ enc, const float* __restrict__ embed,
    const float* __restrict__ x2, const float* __restrict__ e2,
    float* __restrict__ quant, float* __restrict__ diff_acc,
    int B, int D, int K, int HW)
{
  __shared__ alignas(16) unsigned short Alds[64 * 264];
  __shared__ alignas(16) unsigned short Blds[64 * 72];
  __shared__ unsigned long long best[64];
  __shared__ int codes[64];
  __shared__ int pb_s[64], prem_s[64];
  __shared__ float redf[256];
  const int tid = threadIdx.x, wave = tid >> 5, lane = tid & 31;
  const int wm = wave & 3, wn = wave >> 2;
  const unsigned Ntot = (unsigned)B * (unsigned)HW;
  const unsigned nbase = blockIdx.x * 64u;

  // decode the block's 64 positions once
  if (tid < 64) {
    unsigned n = nbase + (unsigned)tid;
    int b = 0, rem = 0;
    if (n < Ntot) {
      b = (int)(n / (unsigned)HW);
      rem = (int)(n - (unsigned)b * (unsigned)HW);
    }
    pb_s[tid] = b; prem_s[tid] = rem;
    best[tid] = ~0ULL;
  }
  __syncthreads();

  // stage all A (64 positions x 256 dims) once; d fixed per thread
  {
    const int d = tid;
    for (int p = 0; p < 64; ++p) {
      unsigned n = nbase + (unsigned)p;
      float v = 0.f;
      if (n < Ntot) v = enc[(pb_s[p] * D + d) * HW + prem_s[p]];
      Alds[p * 264 + d] = f2bf(v);
    }
  }
  __syncthreads();

  const int srow = tid >> 2;
  const int skb  = (tid & 3) * 16;
  v8f zero = {0.f,0.f,0.f,0.f,0.f,0.f,0.f,0.f};

  for (int c0 = 0; c0 < K; c0 += 64) {
    v8f acc[2]; acc[0] = zero; acc[1] = zero;
    for (int k0 = 0; k0 < D; k0 += 64) {
      {
        unsigned short* dst = &Blds[srow * 72 + skb];
        const int code = c0 + srow;
        #pragma unroll
        for (int j = 0; j < 16; ++j) {
          int d = k0 + skb + j;
          dst[j] = f2bf(embed[d * K + code]);
        }
      }
      __syncthreads();
      #pragma unroll
      for (int c = 0; c < 2; ++c) {
        Frag a, b0, b1;
        const unsigned short* ar = &Alds[(wm * 16 + (lane & 15)) * 264 + k0 + c * 32];
        const int kb = (lane < 16) ? 0 : 16;
        a.q[0] = *(const uint4*)(ar + kb);
        a.q[1] = *(const uint4*)(ar + kb + 8);
        const int kc = (lane < 16) ? 0 : 16;
        const unsigned short* br0 = &Blds[(wn * 32 + (lane & 15)) * 72 + c * 32 + kc];
        b0.q[0] = ((const uint4*)br0)[0];
        b0.q[1] = ((const uint4*)br0)[1];
        const unsigned short* br1 = &Blds[(wn * 32 + 16 + (lane & 15)) * 72 + c * 32 + kc];
        b1.q[0] = ((const uint4*)br1)[0];
        b1.q[1] = ((const uint4*)br1)[1];
        acc[0] = __builtin_amdgcn_wmma_f32_16x16x32_bf16(false, a.v, false, b0.v,
                                                         (short)0, acc[0], false, false);
        acc[1] = __builtin_amdgcn_wmma_f32_16x16x32_bf16(false, a.v, false, b1.v,
                                                         (short)0, acc[1], false, false);
      }
      __syncthreads();
    }
    const int m0 = wm * 16 + ((lane >> 4) << 3);
    #pragma unroll
    for (int s = 0; s < 2; ++s) {
      int code = c0 + wn * 32 + s * 16 + (lane & 15);
      float ee = e2[code];
      #pragma unroll
      for (int r = 0; r < 8; ++r) {
        int p = m0 + r;
        unsigned n = nbase + (unsigned)p;
        if (n < Ntot) {
          float dist = x2[n] - 2.f * acc[s][r] + ee;
          unsigned u = __float_as_uint(dist);
          unsigned key = (u & 0x80000000u) ? ~u : (u | 0x80000000u);
          unsigned long long packed = ((unsigned long long)key << 32) | (unsigned)code;
          atomicMin(&best[p], packed);
        }
      }
    }
    __syncthreads();
  }

  if (tid < 64) codes[tid] = (int)(best[tid] & 0xFFFFFFFFu);
  __syncthreads();

  // quant gather (NCHW) + commitment partial; d fixed per thread
  float part = 0.f;
  {
    const int d = tid;
    for (int p = 0; p < 64; ++p) {
      unsigned n = nbase + (unsigned)p;
      if (n < Ntot) {
        int oi = (pb_s[p] * D + d) * HW + prem_s[p];
        float q = embed[d * K + codes[p]];
        float lg = enc[oi];
        float df = q - lg;
        part += df * df;
        quant[oi] = q;
      }
    }
  }
  redf[tid] = part;
  __syncthreads();
  for (int o = 128; o > 0; o >>= 1) {
    if (tid < o) redf[tid] += redf[tid + o];
    __syncthreads();
  }
  if (tid == 0) atomicAdd(diff_acc, redf[0]);
}

__global__ void zero_kernel(float* p) { p[0] = 0.f; }

__global__ void commit_kernel(const float* __restrict__ acc, float* __restrict__ out,
                              float invN) { out[0] = acc[0] * invN; }

__global__ void recon_kernel(const float* __restrict__ img, const float* __restrict__ dec,
                             float* __restrict__ recon, float* __restrict__ outcpy, int n)
{
  int i = blockIdx.x * blockDim.x + threadIdx.x;
  if (i < n) {
    float d = img[i] - dec[i];
    recon[i] = d * d;
    outcpy[i] = dec[i];
  }
}

// ---------------------------------------------------------------------------
// Host orchestration
// ---------------------------------------------------------------------------
extern "C" void kernel_launch(void* const* d_in, const int* in_sizes, int n_in,
                              void* d_out, int out_size, void* d_ws, size_t ws_size,
                              hipStream_t stream)
{
  (void)in_sizes; (void)out_size; (void)ws_size;
  int ii = 0;
  auto P = [&]() -> const float* { return (const float*)d_in[ii++]; };

  char* wsbase = (char*)d_ws;
  size_t off = 0;
  auto alloc = [&](size_t nf) -> float* {
    float* p = (float*)(wsbase + off);
    off += ((nf * sizeof(float) + 255) & ~(size_t)255);
    return p;
  };

  const int Bn = 4;
  float* tmpA = alloc(4ull * 128 * 256 * 256);   // shared GN/SiLU scratch (max tensor)

  auto conv = [&](const float* in, const float* w, const float* b, const float* res,
                  float* out, int Cin, int Hin, int Win, int Cout, int Hout, int Wout,
                  int ks, int pad, int stride, int ups) {
    long npix = (long)Bn * Hout * Wout;
    dim3 grid((unsigned)((npix + 63) / 64), (unsigned)((Cout + 63) / 64));
    if (ks == 3 && ups == 0)
      conv_wmma_kernel<3, 0><<<grid, dim3(256), 0, stream>>>(in, w, b, res, out,
          Bn, Cin, Hin, Win, Cout, Hout, Wout, pad, stride);
    else if (ks == 1)
      conv_wmma_kernel<1, 0><<<grid, dim3(256), 0, stream>>>(in, w, b, res, out,
          Bn, Cin, Hin, Win, Cout, Hout, Wout, pad, stride);
    else
      conv_wmma_kernel<3, 1><<<grid, dim3(256), 0, stream>>>(in, w, b, res, out,
          Bn, Cin, Hin, Win, Cout, Hout, Wout, pad, stride);
  };
  auto gn = [&](const float* in, const float* g, const float* b, float* out,
                int C, int HW, int silu) {
    gn_kernel<<<dim3(Bn * 32), dim3(256), 0, stream>>>(in, g, b, out, C, HW, silu);
  };
  auto cat = [&](const float* a, const float* b2, float* out, int C1, int C2, int HW) {
    long total = (long)Bn * (C1 + C2) * HW;
    concat_kernel<<<dim3((unsigned)((total + 255) / 256)), dim3(256), 0, stream>>>(
        a, b2, out, Bn, C1, C2, HW);
  };
  auto rb = [&](const float* x, float* out, int Cin, int Cout, int H, int W) {
    const float* g1g = P(); const float* g1b = P();
    const float* g2g = P(); const float* g2b = P();
    const float* w1 = P();  const float* b1 = P();
    const float* w2 = P();  const float* b2 = P();
    const float* sw = nullptr; const float* sb = nullptr;
    if (Cin != Cout) { sw = P(); sb = P(); }
    int HW = H * W;
    float* hh = alloc((size_t)Bn * Cout * HW);
    gn(x, g1g, g1b, tmpA, Cin, HW, 1);
    conv(tmpA, w1, b1, nullptr, hh, Cin, H, W, Cout, H, W, 3, 1, 1, 0);
    gn(hh, g2g, g2b, tmpA, Cout, HW, 1);
    const float* res = x;
    if (sw) {
      float* sk = alloc((size_t)Bn * Cout * HW);
      conv(x, sw, sb, nullptr, sk, Cin, H, W, Cout, H, W, 1, 0, 1, 0);
      res = sk;
    }
    conv(tmpA, w2, b2, res, out, Cout, H, W, Cout, H, W, 3, 1, 1, 0);
  };
  auto attn = [&](const float* x, float* out, int C, int H, int W) {
    const float* gg = P(); const float* gb = P();
    const float* qw = P(); const float* qb = P();
    const float* pw = P(); const float* pb = P();
    int T = H * W;
    gn(x, gg, gb, tmpA, C, T, 0);
    float* qkv = alloc((size_t)Bn * 3 * C * T);
    conv(tmpA, qw, qb, nullptr, qkv, C, H, W, 3 * C, H, W, 1, 0, 1, 0);
    float* av = alloc((size_t)Bn * C * T);
    attn_core_kernel<<<dim3(Bn * 4), dim3(256), 0, stream>>>(qkv, av, C, T, 4);
    conv(av, pw, pb, x, out, C, H, W, C, H, W, 1, 0, 1, 0);  // out = proj(a) + x
  };

  const float* img = P();

  // ---------------- encoder ----------------
  float* h0 = alloc(4ull * 64 * 65536);
  { const float* w = P(); const float* b = P();
    conv(img, w, b, nullptr, h0, 3, 256, 256, 64, 256, 256, 3, 1, 1, 0); }
  float* h1 = alloc(4ull * 64 * 16384);
  { const float* w = P(); const float* b = P();
    conv(h0, w, b, nullptr, h1, 64, 256, 256, 64, 128, 128, 3, 1, 2, 0); }
  float* h2 = alloc(4ull * 64 * 4096);
  { const float* w = P(); const float* b = P();
    conv(h1, w, b, nullptr, h2, 64, 128, 128, 64, 64, 64, 3, 1, 2, 0); }
  float* h3 = alloc(4ull * 64 * 1024);
  { const float* w = P(); const float* b = P();
    conv(h2, w, b, nullptr, h3, 64, 64, 64, 64, 32, 32, 3, 1, 2, 0); }
  float* m1 = alloc(4ull * 64 * 1024);  rb(h3, m1, 64, 64, 32, 32);
  float* ma = alloc(4ull * 64 * 1024);  attn(m1, ma, 64, 32, 32);
  float* m2 = alloc(4ull * 64 * 1024);  rb(ma, m2, 64, 64, 32, 32);
  float* c0 = alloc(4ull * 128 * 1024); cat(m2, h3, c0, 64, 64, 1024);
  float* u0 = alloc(4ull * 512 * 1024); rb(c0, u0, 128, 512, 32, 32);
  float* u0u = alloc(4ull * 512 * 4096);
  { const float* w = P(); const float* b = P();
    conv(u0, w, b, nullptr, u0u, 512, 32, 32, 512, 64, 64, 3, 1, 1, 1); }
  float* c1 = alloc(4ull * 576 * 4096); cat(u0u, h2, c1, 512, 64, 4096);
  float* u1 = alloc(4ull * 256 * 4096); rb(c1, u1, 576, 256, 64, 64);
  float* enc_out = alloc(4ull * 256 * 4096);
  { const float* gg = P(); const float* gb = P();
    gn(u1, gg, gb, tmpA, 256, 4096, 1);
    const float* w = P(); const float* b = P();
    conv(tmpA, w, b, nullptr, enc_out, 256, 64, 64, 256, 64, 64, 3, 1, 1, 0); }

  // ---------------- vector quantize ----------------
  const float* embed = (const float*)d_in[n_in - 1];
  float* x2 = alloc(16384);
  float* e2 = alloc(8192);
  float* quant = alloc(4ull * 256 * 4096);
  float* diffacc = alloc(64);
  zero_kernel<<<dim3(1), dim3(1), 0, stream>>>(diffacc);
  rowsq_kernel<<<dim3(64), dim3(256), 0, stream>>>(enc_out, x2, 4, 256, 4096);
  colsq_kernel<<<dim3(32), dim3(256), 0, stream>>>(embed, e2, 256, 8192);
  vq_kernel<<<dim3(256), dim3(256), 0, stream>>>(enc_out, embed, x2, e2,
                                                 quant, diffacc, 4, 256, 8192, 4096);

  // ---------------- decoder ----------------
  float* g0 = alloc(4ull * 64 * 4096);
  { const float* w = P(); const float* b = P();
    conv(quant, w, b, nullptr, g0, 256, 64, 64, 64, 64, 64, 3, 1, 1, 0); }
  float* g1 = alloc(4ull * 64 * 1024);
  { const float* w = P(); const float* b = P();
    conv(g0, w, b, nullptr, g1, 64, 64, 64, 64, 32, 32, 3, 1, 2, 0); }
  float* dm1 = alloc(4ull * 64 * 1024);  rb(g1, dm1, 64, 64, 32, 32);
  float* dma = alloc(4ull * 64 * 1024);  attn(dm1, dma, 64, 32, 32);
  float* dm2 = alloc(4ull * 64 * 1024);  rb(dma, dm2, 64, 64, 32, 32);
  float* dc0 = alloc(4ull * 128 * 1024); cat(dm2, g1, dc0, 64, 64, 1024);
  float* du0 = alloc(4ull * 512 * 1024); rb(dc0, du0, 128, 512, 32, 32);
  float* du0u = alloc(4ull * 512 * 4096);
  { const float* w = P(); const float* b = P();
    conv(du0, w, b, nullptr, du0u, 512, 32, 32, 512, 64, 64, 3, 1, 1, 1); }
  float* dc1 = alloc(4ull * 576 * 4096); cat(du0u, g0, dc1, 512, 64, 4096);
  float* du1 = alloc(4ull * 256 * 4096); rb(dc1, du1, 576, 256, 64, 64);
  float* du1u = alloc(4ull * 256 * 16384);
  { const float* w = P(); const float* b = P();
    conv(du1, w, b, nullptr, du1u, 256, 64, 64, 256, 128, 128, 3, 1, 1, 1); }
  float* du2 = alloc(4ull * 128 * 16384); rb(du1u, du2, 256, 128, 128, 128);
  float* du2u = alloc(4ull * 128 * 65536);
  { const float* w = P(); const float* b = P();
    conv(du2, w, b, nullptr, du2u, 128, 128, 128, 128, 256, 256, 3, 1, 1, 1); }
  float* du3 = alloc(4ull * 64 * 65536); rb(du2u, du3, 128, 64, 256, 256);
  float* dec_out = alloc(4ull * 3 * 65536);
  { const float* gg = P(); const float* gb = P();
    gn(du3, gg, gb, tmpA, 64, 65536, 1);
    const float* w = P(); const float* b = P();
    conv(tmpA, w, b, nullptr, dec_out, 64, 256, 256, 3, 256, 256, 3, 1, 1, 0); }

  // ---------------- outputs: [recon(786432), commit(1), out(786432)] ----------------
  float* fout = (float*)d_out;
  const int nrec = 4 * 3 * 65536;
  recon_kernel<<<dim3((unsigned)((nrec + 255) / 256)), dim3(256), 0, stream>>>(
      img, dec_out, fout, fout + nrec + 1, nrec);
  commit_kernel<<<dim3(1), dim3(1), 0, stream>>>(diffacc, fout + nrec,
                                                 1.f / (16384.f * 256.f));
}
